// FCOS_73787538145418
// MI455X (gfx1250) — matrix operand order repaired and checked by reading, not compiled, not run on hardware
//
#include <hip/hip_runtime.h>

// ---------------------------------------------------------------------------
// Types for WMMA fragments (wave32, gfx1250)
// ---------------------------------------------------------------------------
typedef __bf16 v8bf  __attribute__((ext_vector_type(8)));
typedef __bf16 v16bf __attribute__((ext_vector_type(16)));
typedef float  v8f   __attribute__((ext_vector_type(8)));

// A-fragment: lanes hold K {kA..kA+15} (contiguous); p already includes kA
__device__ __forceinline__ v16bf loadA16(const __bf16* p) {
    v8bf lo = *(const v8bf*)(p);
    v8bf hi = *(const v8bf*)(p + 8);
    return __builtin_shufflevector(lo, hi,
        0, 1, 2, 3, 4, 5, 6, 7, 8, 9, 10, 11, 12, 13, 14, 15);
}

// Row index (past every level's data) of the zeroed padding row kept inside
// every activation buffer; buffers are allocated with one extra row.
#define ZROWI (8 * 64 * 64)

// ---------------------------------------------------------------------------
// Layout transforms
// ---------------------------------------------------------------------------
__global__ void k_nchw_to_nhwc(const float* __restrict__ in, __bf16* __restrict__ out,
                               int B, int C, int H, int W) {
    size_t i = (size_t)blockIdx.x * blockDim.x + threadIdx.x;
    size_t total = (size_t)B * C * H * W;
    if (i >= total) return;
    int c = (int)(i % C);
    size_t r = i / C;
    int w = (int)(r % W); r /= W;
    int h = (int)(r % H);
    int b = (int)(r / H);
    out[i] = (__bf16)in[(((size_t)b * C + c) * H + h) * W + w];
}

// fp32 OIHW (3x3) -> bf16 [tap][OP][I], writing channels [coOff, coOff+O)
__global__ void k_wxform(const float* __restrict__ w, __bf16* __restrict__ out,
                         int O, int I, int OP, int coOff) {
    size_t i = (size_t)blockIdx.x * blockDim.x + threadIdx.x;
    size_t total = (size_t)9 * O * I;
    if (i >= total) return;
    int ci = (int)(i % I);
    size_t r = i / I;
    int co  = (int)(r % O);
    int tap = (int)(r / O);
    out[((size_t)tap * OP + coOff + co) * I + ci] =
        (__bf16)w[((size_t)co * I + ci) * 9 + tap];
}

__global__ void k_zero_bf16(__bf16* p, size_t n) {
    size_t i = (size_t)blockIdx.x * blockDim.x + threadIdx.x;
    if (i < n) p[i] = (__bf16)0.0f;
}
__global__ void k_zero_f32(float* p, int n) {
    int i = blockIdx.x * blockDim.x + threadIdx.x;
    if (i < n) p[i] = 0.0f;
}
__global__ void k_copy_f32(const float* __restrict__ in, float* __restrict__ out,
                           int n, int off) {
    int i = blockIdx.x * blockDim.x + threadIdx.x;
    if (i < n) out[off + i] = in[i];
}

// ---------------------------------------------------------------------------
// 3x3 'SAME' conv as implicit GEMM on v_wmma_f32_16x16x32_bf16.
//   X  : bf16 NHWC activations, Cin = 256;  Wt : bf16 [9][CoutP][256]
//   Wave tile: 32M x 64N (8 f32 accumulators), block = 8 waves = 256M x 64N.
//   K pipeline: double-buffered 32-wide chunks (prefetch next while 8 WMMAs run).
//   Padding: invalid lanes read row ZROWI (zeroed) of the SAME buffer.
//   All addresses are straight GEPs off kernarg bases recomputed per tap
//   (no loop-carried pointer PHIs, no pointer selects) so address-space
//   inference lowers every access to global_load/global_store.
// PRED=false: Y = bf16 NHWC (stride CoutP), optional ReLU.
// PRED=true : fp32 into packed rows (b*totalHW + lvlOff + hw)*25, ch [chOff,+chLim).
// ---------------------------------------------------------------------------
template <bool RELU, bool PRED>
__global__ __launch_bounds__(256)
void conv3x3_bf16_wmma(const __bf16* __restrict__ Xb_, ptrdiff_t dX,
                       const __bf16* __restrict__ Wb_, ptrdiff_t dW,
                       const float*  __restrict__ Bb_, ptrdiff_t dB,
                       __bf16* __restrict__ Yb_, ptrdiff_t dY,
                       float* __restrict__ Yf,
                       int H, int W, int CoutP, int totalHW, int lvlOff,
                       int chOff0, int chLim0, int chOff1, int chLim1) {
    const int z = blockIdx.z;
    const __bf16* X    = Xb_ + (z ? dX : 0);
    const __bf16* Wt   = Wb_ + (z ? dW : 0);
    const float*  bias = Bb_ + (z ? dB : 0);
    __bf16*       Y    = Yb_ + (z ? dY : 0);
    const int chOff = z ? chOff1 : chOff0;
    const int chLim = z ? chLim1 : chLim0;

    const int lane = threadIdx.x & 31;
    const int wave = threadIdx.x >> 5;
    const int mt   = (blockIdx.x * 8 + wave) * 32;   // 32 M rows per wave
    const int n0   = blockIdx.y * 64;
    const int HW   = H * W;

    const int lm = lane & 15;
    const int kA = (lane & 16) ? 16 : 0;             // A K half
    const int kB = (lane & 16) ? 16 : 0;             // B K half

    const int m0 = mt + lm;                          // M-fragment 0 row
    const int m1 = mt + 16 + lm;                     // M-fragment 1 row
    const int b0 = m0 / HW, hw0 = m0 - b0 * HW;
    const int h0 = hw0 / W, w0 = hw0 - h0 * W;
    const int b1 = m1 / HW, hw1 = m1 - b1 * HW;
    const int h1 = hw1 / W, w1 = hw1 - h1 * W;

    v8f acc[8];
#pragma unroll
    for (int j = 0; j < 8; ++j)
#pragma unroll
        for (int r = 0; r < 8; ++r) acc[j][r] = 0.0f;

    // per-j weight ROW OFFSETS (integers, include kB) -- no pointer PHIs
    size_t wrow[4];
#pragma unroll
    for (int j = 0; j < 4; ++j)
        wrow[j] = (size_t)(n0 + j * 16 + lm) * 256 + kB;
    const size_t wtapStride = (size_t)CoutP * 256;

    int dh = -1, dw = -1;
#pragma unroll 1
    for (int tap = 0; tap < 9; ++tap) {
        const bool v0 = (h0 + dh >= 0) & (h0 + dh < H) & (w0 + dw >= 0) & (w0 + dw < W);
        const bool v1 = (h1 + dh >= 0) & (h1 + dh < H) & (w1 + dw >= 0) & (w1 + dw < W);
        const int  moff = dh * W + dw;               // neighbor offset in NHWC rows
        // single-base addressing: select the ROW INDEX, not the pointer
        const __bf16* a0 = X + (size_t)(v0 ? (m0 + moff) : ZROWI) * 256 + kA;
        const __bf16* a1 = X + (size_t)(v1 ? (m1 + moff) : ZROWI) * 256 + kA;
        // weight base recomputed from the kernarg pointer every tap
        const __bf16* wtap = Wt + (size_t)tap * wtapStride;

        // double-buffered K pipeline over 8 chunks of 32
        v16bf Ab[2][2];          // [buf][mfrag]
        v16bf Bb[2][4];          // [buf][j]
        Ab[0][0] = loadA16(a0);
        Ab[0][1] = loadA16(a1);
#pragma unroll
        for (int j = 0; j < 4; ++j) Bb[0][j] = *(const v16bf*)(wtap + wrow[j]);

#pragma unroll
        for (int kc = 0; kc < 8; ++kc) {
            const int cur = kc & 1, nxt = cur ^ 1;
            if (kc < 7) {
                const int kb = (kc + 1) * 32;
                Ab[nxt][0] = loadA16(a0 + kb);
                Ab[nxt][1] = loadA16(a1 + kb);
#pragma unroll
                for (int j = 0; j < 4; ++j)
                    Bb[nxt][j] = *(const v16bf*)(wtap + wrow[j] + kb);
            }
#pragma unroll
            for (int j = 0; j < 4; ++j)
                acc[j] = __builtin_amdgcn_wmma_f32_16x16x32_bf16(
                    false, Ab[cur][0], false, Bb[cur][j], (short)0, acc[j], false, false);
#pragma unroll
            for (int j = 0; j < 4; ++j)
                acc[4 + j] = __builtin_amdgcn_wmma_f32_16x16x32_bf16(
                    false, Ab[cur][1], false, Bb[cur][j], (short)0, acc[4 + j], false, false);
        }

        if (++dw > 1) { dw = -1; ++dh; }             // incremental (dh,dw), no div/mod
    }

    // Epilogue: bias (+ReLU) and store per C/D VGPR layout
#pragma unroll
    for (int f = 0; f < 2; ++f) {
        const int mrb = mt + f * 16 + ((lane & 16) ? 8 : 0);
#pragma unroll
        for (int j = 0; j < 4; ++j) {
            const int nn = j * 16 + lm;
            const float bv = bias[n0 + nn];
#pragma unroll
            for (int r = 0; r < 8; ++r) {
                const int mr = mrb + r;
                float v = acc[f * 4 + j][r] + bv;
                if (RELU) v = fmaxf(v, 0.0f);
                if (PRED) {
                    if (nn < chLim) {
                        const int bb  = mr / HW;
                        const int ohw = mr - bb * HW;
                        Yf[((size_t)bb * totalHW + lvlOff + ohw) * 25 + chOff + nn] = v;
                    }
                } else {
                    Y[(size_t)mr * CoutP + (n0 + nn)] = (__bf16)v;
                }
            }
        }
    }
}

// ---------------------------------------------------------------------------
// Host orchestration
// ---------------------------------------------------------------------------
extern "C" void kernel_launch(void* const* d_in, const int* in_sizes, int n_in,
                              void* d_out, int out_size, void* d_ws, size_t ws_size,
                              hipStream_t stream) {
    (void)in_sizes; (void)n_in; (void)out_size; (void)ws_size;

    const float* p3   = (const float*)d_in[0];
    const float* p4   = (const float*)d_in[1];
    const float* p5   = (const float*)d_in[2];
    const float* scw  = (const float*)d_in[3];   // (4,256,256,3,3)
    const float* scb  = (const float*)d_in[4];   // (4,256)
    const float* sbw  = (const float*)d_in[5];
    const float* sbb  = (const float*)d_in[6];
    const float* pcw  = (const float*)d_in[7];   // (20,256,3,3)
    const float* pcb  = (const float*)d_in[8];
    const float* pbw  = (const float*)d_in[9];   // (4,256,3,3)
    const float* pbb  = (const float*)d_in[10];
    const float* pctw = (const float*)d_in[11];  // (1,256,3,3)
    const float* pctb = (const float*)d_in[12];
    float* out = (float*)d_out;

    char*  ws  = (char*)d_ws;
    size_t off = 0;
    auto alloc = [&](size_t bytes) -> char* {
        off = (off + 255) & ~(size_t)255;
        char* p = ws + off;
        off += bytes;
        return p;
    };

    const size_t ACT  = (size_t)8 * 64 * 64 * 256;   // max-level NHWC elems
    const size_t ACTA = ACT + 256;                   // + zeroed padding row (ZROWI)
    __bf16* Xb = (__bf16*)alloc(ACTA * 2);
    __bf16* C0 = (__bf16*)alloc(ACTA * 2);
    __bf16* C1 = (__bf16*)alloc(ACTA * 2);
    __bf16* B0 = (__bf16*)alloc(ACTA * 2);
    __bf16* B1 = (__bf16*)alloc(ACTA * 2);
    const size_t SWL = (size_t)9 * 256 * 256;        // one stem layer, xformed
    __bf16* SW  = (__bf16*)alloc(8 * SWL * 2);       // 4 cls + 4 box layers
    const size_t PWN = (size_t)9 * 64 * 256;
    __bf16* PWc = (__bf16*)alloc(PWN * 2);           // cls head, padded to 64
    __bf16* PWb = (__bf16*)alloc(PWN * 2);           // box(0..3)+ctr(4), padded
    float*  SB  = (float*)alloc(2048 * 4);           // stem biases: cls[4][256], box[4][256]
    float*  PBc = (float*)alloc(64 * 4);
    float*  PBb = (float*)alloc(64 * 4);

    // --- weight / bias transforms + zero pad rows ---
    {
        int thr = 256;
        int g = (int)((SWL + thr - 1) / thr);
        for (int i = 0; i < 4; ++i) {
            k_wxform<<<g, thr, 0, stream>>>(scw + (size_t)i * 256 * 256 * 9,
                                            SW + (size_t)i * SWL, 256, 256, 256, 0);
            k_wxform<<<g, thr, 0, stream>>>(sbw + (size_t)i * 256 * 256 * 9,
                                            SW + (size_t)(4 + i) * SWL, 256, 256, 256, 0);
        }
        int gz = (int)((PWN + thr - 1) / thr);
        k_zero_bf16<<<gz, thr, 0, stream>>>(PWc, PWN);
        k_zero_bf16<<<gz, thr, 0, stream>>>(PWb, PWN);
        // zero the padding row of every activation buffer
        k_zero_bf16<<<1, 256, 0, stream>>>(Xb + ACT, 256);
        k_zero_bf16<<<1, 256, 0, stream>>>(C0 + ACT, 256);
        k_zero_bf16<<<1, 256, 0, stream>>>(C1 + ACT, 256);
        k_zero_bf16<<<1, 256, 0, stream>>>(B0 + ACT, 256);
        k_zero_bf16<<<1, 256, 0, stream>>>(B1 + ACT, 256);
        k_zero_f32<<<1, 64, 0, stream>>>(PBc, 64);
        k_zero_f32<<<1, 64, 0, stream>>>(PBb, 64);
        // stem biases into one workspace buffer: cls at [0..1023], box at [1024..2047]
        k_copy_f32<<<4, 256, 0, stream>>>(scb, SB, 1024, 0);
        k_copy_f32<<<4, 256, 0, stream>>>(sbb, SB, 1024, 1024);
        k_wxform<<<(9 * 20 * 256 + thr - 1) / thr, thr, 0, stream>>>(pcw,  PWc, 20, 256, 64, 0);
        k_wxform<<<(9 * 4  * 256 + thr - 1) / thr, thr, 0, stream>>>(pbw,  PWb, 4,  256, 64, 0);
        k_wxform<<<(9 * 1  * 256 + thr - 1) / thr, thr, 0, stream>>>(pctw, PWb, 1,  256, 64, 4);
        k_copy_f32<<<1, 32, 0, stream>>>(pcb,  PBc, 20, 0);
        k_copy_f32<<<1, 32, 0, stream>>>(pbb,  PBb, 4,  0);
        k_copy_f32<<<1, 32, 0, stream>>>(pctb, PBb, 1,  4);
    }

    // --- per-FPN-level head ---
    struct Lvl { const float* x; int H, W, lvlOff; };
    Lvl lv[3] = { {p3, 64, 64, 0}, {p4, 32, 32, 4096}, {p5, 16, 16, 5120} };
    const int totalHW = 5376;

    for (int L = 0; L < 3; ++L) {
        const int H = lv[L].H, W = lv[L].W;
        const size_t N = (size_t)8 * 256 * H * W;
        k_nchw_to_nhwc<<<(int)((N + 255) / 256), 256, 0, stream>>>(lv[L].x, Xb, 8, 256, H, W);

        const int Mtiles32 = 8 * H * W / 32;         // 32-row wave tiles; mult. of 8
        dim3 gs(Mtiles32 / 8, 4, 2);                 // stems: Cout=256, z = cls/box
        dim3 gp(Mtiles32 / 8, 1, 2);                 // preds: CoutP=64, z = cls/box+ctr

        // fused cls+box stem chains: X -> {C0,B0} -> {C1,B1} -> {C0,B0} -> {C1,B1}
        const __bf16* srcC = Xb;
        const __bf16* srcB = Xb;
        __bf16* cdst[4] = { C0, C1, C0, C1 };
        __bf16* bdst[4] = { B0, B1, B0, B1 };
        for (int i = 0; i < 4; ++i) {
            conv3x3_bf16_wmma<true, false><<<gs, 256, 0, stream>>>(
                srcC, (ptrdiff_t)(srcB - srcC),
                SW + (size_t)i * SWL, (ptrdiff_t)(4 * SWL),
                SB + (size_t)i * 256, (ptrdiff_t)1024,
                cdst[i], (ptrdiff_t)(bdst[i] - cdst[i]),
                nullptr,
                H, W, 256, 0, 0, 0, 0, 0, 0);
            srcC = cdst[i];
            srcB = bdst[i];
        }

        // fused prediction heads write straight into the packed (B, 5376, 25) output
        conv3x3_bf16_wmma<false, true><<<gp, 256, 0, stream>>>(
            srcC, (ptrdiff_t)(srcB - srcC),
            PWc, (ptrdiff_t)(PWb - PWc),
            PBc, (ptrdiff_t)(PBb - PBc),
            C0, (ptrdiff_t)0,            // unused store base (never written, PRED)
            out,
            H, W, 64, totalHW, lv[L].lvlOff, 0, 20, 20, 5);
    }
}
